// EncoderDecoderCDE_59897613910627
// MI455X (gfx1250) — compile-verified, matrix operand order checked by reading, not run on hardware
//
#include <hip/hip_runtime.h>
#include <hip/hip_bf16.h>
#include <math.h>
#include <stdint.h>

// ---------------------------------------------------------------------------
// CDNA5 (gfx1250) WMMA types and fragment loader.
// 16-bit A (16x32 MxK) / B (32x16 KxN, stored as W row-major NxK for X@W^T):
//   lane<16  -> row = lane,    K elements {kc+0..7,  kc+16..23}
//   lane>=16 -> row = lane-16, K elements {kc+8..15, kc+24..31}
// i.e. two contiguous 16-byte loads at (kc + kbase) and (kc + kbase + 16).
// ---------------------------------------------------------------------------
typedef __attribute__((ext_vector_type(16))) _Float16 v16h;
typedef __attribute__((ext_vector_type(8)))  _Float16 v8h;
typedef __attribute__((ext_vector_type(8)))  float    v8f;

union ABFrag { v16h v; v8h h[2]; };

__device__ __forceinline__ v16h load_frag(const _Float16* p) {
  ABFrag f;
  f.h[0] = *(const v8h*)(p);
  f.h[1] = *(const v8h*)(p + 16);
  return f.v;
}

__device__ __forceinline__ v8f wmma_f16(v16h a, v16h b, v8f c) {
  return __builtin_amdgcn_wmma_f32_16x16x32_f16(false, a, false, b, (short)0, c,
                                                false, false);
}

__device__ __forceinline__ float sigm(float x) { return 1.f / (1.f + expf(-x)); }

// ---------------------------------------------------------------------------
// CDNA5 async memory->LDS copy (ASYNCcnt-tracked), CDNA5 ISA.
// One instruction moves 16B per lane. Completion is in-order, so
// s_wait_asynccnt 8 after issuing the next 8-op stage == previous stage done.
// ---------------------------------------------------------------------------
__device__ __forceinline__ void async_load_b128(uint32_t lds_addr,
                                                const void* gaddr) {
  asm volatile("global_load_async_to_lds_b128 %0, %1, off"
               :: "v"(lds_addr), "v"((unsigned long long)(uintptr_t)gaddr)
               : "memory");
}
__device__ __forceinline__ void wait_async_le8() {
  asm volatile("s_wait_asynccnt 0x8" ::: "memory");
}
__device__ __forceinline__ void wait_async_0() {
  asm volatile("s_wait_asynccnt 0x0" ::: "memory");
}

enum { ACT_NONE = 0, ACT_RELU = 1, ACT_TANH = 2 };

// ---------------------------------------------------------------------------
// Generic WMMA GEMM: out[M,N] = act(A[M,K] @ W[N,K]^T + bias[N])
// One wave per 64x16 C stripe (4 M-tiles): the B fragment is loaded once per
// 32-K step and feeds 4 WMMAs (4x W-traffic reuse). M % 64 == 0, K % 32 == 0.
// ---------------------------------------------------------------------------
template <int ACT, bool OUT_F16>
__global__ void wmma_gemm(const _Float16* __restrict__ A,
                          const _Float16* __restrict__ W,
                          const float* __restrict__ bias,
                          void* __restrict__ out, int M, int N, int K) {
  const int ntN  = N >> 4;
  const int tile = blockIdx.x;
  const int mb   = tile / ntN;          // 64-row block
  const int nt   = tile - mb * ntN;     // 16-col tile
  const int lane = threadIdx.x & 31;
  const int l16  = lane & 15;
  const int kb   = (lane >> 4) << 3;    // 0 or 8
  const size_t arow = (size_t)(mb * 64 + l16) * K;
  const size_t wrow = (size_t)(nt * 16 + l16) * K;

  v8f acc0 = {}, acc1 = {}, acc2 = {}, acc3 = {};
  for (int kc = 0; kc < K; kc += 32) {
    v16h b  = load_frag(W + wrow + kc + kb);
    v16h x0 = load_frag(A + arow + (size_t)0 * 16 * K + kc + kb);
    v16h x1 = load_frag(A + arow + (size_t)1 * 16 * K + kc + kb);
    v16h x2 = load_frag(A + arow + (size_t)2 * 16 * K + kc + kb);
    v16h x3 = load_frag(A + arow + (size_t)3 * 16 * K + kc + kb);
    acc0 = wmma_f16(x0, b, acc0);
    acc1 = wmma_f16(x1, b, acc1);
    acc2 = wmma_f16(x2, b, acc2);
    acc3 = wmma_f16(x3, b, acc3);
  }

  const int col  = nt * 16 + l16;
  const float bv = bias ? bias[col] : 0.f;
  const int mofs = (lane >> 4) << 3;    // C tile: lanes>=16 hold M = r+8
  v8f accs[4] = {acc0, acc1, acc2, acc3};
#pragma unroll
  for (int s = 0; s < 4; ++s) {
#pragma unroll
    for (int r = 0; r < 8; ++r) {
      const int m = mb * 64 + s * 16 + mofs + r;
      float v = accs[s][r] + bv;
      if (ACT == ACT_RELU) v = fmaxf(v, 0.f);
      if (ACT == ACT_TANH) v = tanhf(v);
      if (OUT_F16) ((_Float16*)out)[(size_t)m * N + col] = (_Float16)v;
      else         ((float*)out)[(size_t)m * N + col]    = v;
    }
  }
}

// ---------------------------------------------------------------------------
// Fused LSTM recurrent step: g = gx[t] + h_prev@Whh^T (WMMA, 4 gates share the
// A fragment), then gate math + cell update in-register. grid = (B/16, H/16).
// ---------------------------------------------------------------------------
__global__ void lstm_step(const _Float16* __restrict__ hprev,   // B x 128
                          const _Float16* __restrict__ Whh,     // 512 x 128
                          const _Float16* __restrict__ gx,      // pre-offset by t; row stride gx_rs
                          size_t gx_rs,
                          float* __restrict__ cst,              // B x 128 (in/out)
                          _Float16* __restrict__ hout,          // B x 128
                          _Float16* __restrict__ seqout,        // pre-offset by t, stride seq_rs (or null)
                          size_t seq_rs) {
  const int mt   = blockIdx.x;
  const int nt   = blockIdx.y;
  const int lane = threadIdx.x & 31;
  const int l16  = lane & 15;
  const int kb   = (lane >> 4) << 3;
  const int u    = nt * 16 + l16;
  const size_t arow = (size_t)(mt * 16 + l16) * 128;

  v8f acc0 = {}, acc1 = {}, acc2 = {}, acc3 = {};
#pragma unroll
  for (int kc = 0; kc < 128; kc += 32) {
    v16h a  = load_frag(hprev + arow + kc + kb);
    v16h w0 = load_frag(Whh + (size_t)(0 * 128 + u) * 128 + kc + kb);
    v16h w1 = load_frag(Whh + (size_t)(1 * 128 + u) * 128 + kc + kb);
    v16h w2 = load_frag(Whh + (size_t)(2 * 128 + u) * 128 + kc + kb);
    v16h w3 = load_frag(Whh + (size_t)(3 * 128 + u) * 128 + kc + kb);
    acc0 = wmma_f16(a, w0, acc0);
    acc1 = wmma_f16(a, w1, acc1);
    acc2 = wmma_f16(a, w2, acc2);
    acc3 = wmma_f16(a, w3, acc3);
  }

  const int mofs = (lane >> 4) << 3;
#pragma unroll
  for (int r = 0; r < 8; ++r) {
    const int b = mt * 16 + mofs + r;
    const _Float16* g = gx + (size_t)b * gx_rs;
    const float gi = acc0[r] + (float)g[0 * 128 + u];
    const float gf = acc1[r] + (float)g[1 * 128 + u];
    const float gg = acc2[r] + (float)g[2 * 128 + u];
    const float go = acc3[r] + (float)g[3 * 128 + u];
    const float cold = cst[(size_t)b * 128 + u];
    const float cn = sigm(gf) * cold + sigm(gi) * tanhf(gg);
    const float hn = sigm(go) * tanhf(cn);
    cst[(size_t)b * 128 + u]  = cn;
    hout[(size_t)b * 128 + u] = (_Float16)hn;
    if (seqout) seqout[(size_t)b * seq_rs + u] = (_Float16)hn;
  }
}

// ---------------------------------------------------------------------------
// LayerNorm over 64 dims -> f16 (WMMA A input). One block (64 thr) per row.
// ---------------------------------------------------------------------------
__global__ void layernorm64(const float* __restrict__ z,
                            const float* __restrict__ gamma,
                            const float* __restrict__ beta,
                            _Float16* __restrict__ out) {
  __shared__ float s[64];
  const int b = blockIdx.x, t = threadIdx.x;
  const float v = z[(size_t)b * 64 + t];
  s[t] = v; __syncthreads();
  for (int o = 32; o > 0; o >>= 1) { if (t < o) s[t] += s[t + o]; __syncthreads(); }
  const float mu = s[0] * (1.f / 64.f); __syncthreads();
  const float d = v - mu;
  s[t] = d * d; __syncthreads();
  for (int o = 32; o > 0; o >>= 1) { if (t < o) s[t] += s[t + o]; __syncthreads(); }
  const float rs = rsqrtf(s[0] * (1.f / 64.f) + 1e-5f);
  out[(size_t)b * 64 + t] = (_Float16)(d * rs * gamma[t] + beta[t]);
}

// ---------------------------------------------------------------------------
// Fused MLP2 + einsum: kout[b,h] = sum_i tanh(a@W2^T + b2)[b, h*33+i] * dx[b,i]
// Block = 4 waves owning one 16-row stripe. A fragments live in registers.
// W2 tiles are staged into wave-private LDS double buffers with
// global_load_async_to_lds_b128, overlapping copy of tile nt+4 with the 4
// WMMAs of tile nt; fragments are read back via ds_load_b128. Accumulation
// into a 16x64 LDS tile via ds_add_f32 atomics.
// ---------------------------------------------------------------------------
__global__ void vf_mlp2(const _Float16* __restrict__ A,    // 1024 x 128 f16
                        const _Float16* __restrict__ W2,   // 2112 x 128 f16
                        const float* __restrict__ b2,      // 2112
                        const float* __restrict__ dx,      // 1024 x 33 (this time)
                        float* __restrict__ kout) {        // 1024 x 64
  __shared__ float accum[16 * 64];
  __shared__ __attribute__((aligned(16))) _Float16 wbuf[4][2][16 * 128]; // 32KB
  const int lane = threadIdx.x & 31;
  const int wave = threadIdx.x >> 5;
  const int l16  = lane & 15;
  const int kb   = (lane >> 4) << 3;
  const int mofs = (lane >> 4) << 3;
  const size_t arow = (size_t)(blockIdx.x * 16 + l16) * 128;

  const v16h a0 = load_frag(A + arow +  0 + kb);
  const v16h a1 = load_frag(A + arow + 32 + kb);
  const v16h a2 = load_frag(A + arow + 64 + kb);
  const v16h a3 = load_frag(A + arow + 96 + kb);

  for (int i = threadIdx.x; i < 16 * 64; i += blockDim.x) accum[i] = 0.f;
  __syncthreads();

  // async-stage one 16x128 W2 tile (4KB) into this wave's LDS buffer:
  // 8 ops x (16B per lane x 32 lanes) = 4096B, lane-contiguous layout.
  auto stage = [&](int buf, int nt) {
    const char* g = (const char*)(W2 + (size_t)nt * 16 * 128);
    const uint32_t lb = (uint32_t)(uintptr_t)&wbuf[wave][buf][0];
#pragma unroll
    for (int c = 0; c < 8; ++c) {
      const int byte = c * 512 + lane * 16;
      async_load_b128(lb + byte, g + byte);
    }
  };

  int buf = 0;
  stage(buf, wave);                       // first tile for this wave
  for (int nt = wave; nt < 132; nt += 4) {
    if (nt + 4 < 132) { stage(buf ^ 1, nt + 4); wait_async_le8(); }
    else              { wait_async_0(); }

    const _Float16* wt = &wbuf[wave][buf][0];
    const size_t wr = (size_t)l16 * 128;  // row inside staged tile
    v8f acc = {};
    acc = wmma_f16(a0, load_frag(wt + wr +  0 + kb), acc);
    acc = wmma_f16(a1, load_frag(wt + wr + 32 + kb), acc);
    acc = wmma_f16(a2, load_frag(wt + wr + 64 + kb), acc);
    acc = wmma_f16(a3, load_frag(wt + wr + 96 + kb), acc);

    const int col = nt * 16 + l16;
    const float bv = b2[col];
    const int h  = col / 33;
    const int ic = col - h * 33;
#pragma unroll
    for (int r = 0; r < 8; ++r) {
      const int m = mofs + r;
      const int b = blockIdx.x * 16 + m;
      const float v = tanhf(acc[r] + bv);
      atomicAdd(&accum[m * 64 + h], v * dx[(size_t)b * 33 + ic]);
    }
    buf ^= 1;
  }
  __syncthreads();
  for (int i = threadIdx.x; i < 16 * 64; i += blockDim.x) {
    const int m = i >> 6, hh = i & 63;
    kout[(size_t)(blockIdx.x * 16 + m) * 64 + hh] = accum[i];
  }
}

// ---------------------------------------------------------------------------
// dXdt precompute for all 96 (step,stage) times: dx[ti][b][ch]
// ---------------------------------------------------------------------------
__global__ void compute_dx(const float* __restrict__ coeffs,  // B x 4 x 4 x 33
                           float* __restrict__ dx, int total) {
  const int i = blockIdx.x * blockDim.x + threadIdx.x;
  if (i >= total) return;
  const int ch  = i % 33;
  const int rem = i / 33;
  const int b   = rem & 1023;
  const int ti  = rem >> 10;
  const int step = ti / 6, s = ti - step * 6;
  const float cs[6] = {0.f, 0.2f, 0.3f, 0.8f, 8.f / 9.f, 1.f};
  const float dt = 1.f / 16.f, seg = 0.25f;
  const float t = ((float)step + cs[s]) * dt;
  int idx = (int)floorf(t / seg);
  idx = idx < 0 ? 0 : (idx > 3 ? 3 : idx);
  const float u = t - (float)idx * seg;
  const float* c = coeffs + (((size_t)b * 4 + idx) * 4) * 33;
  dx[i] = c[1 * 33 + ch] + 2.f * c[2 * 33 + ch] * u + 3.f * c[3 * 33 + ch] * u * u;
}

// ---------------------------------------------------------------------------
// Elementwise helpers
// ---------------------------------------------------------------------------
__global__ void axpy6(float* __restrict__ out, const float* __restrict__ z,
                      const float* p1, float c1, const float* p2, float c2,
                      const float* p3, float c3, const float* p4, float c4,
                      const float* p5, float c5, const float* p6, float c6, int n) {
  const int i = blockIdx.x * blockDim.x + threadIdx.x;
  if (i < n)
    out[i] = z[i] + c1 * p1[i] + c2 * p2[i] + c3 * p3[i] + c4 * p4[i] +
             c5 * p5[i] + c6 * p6[i];
}

__global__ void cvt_f16(_Float16* dst, const float* src, size_t n) {
  const size_t i = (size_t)blockIdx.x * blockDim.x + threadIdx.x;
  if (i < n) dst[i] = (_Float16)src[i];
}

__global__ void vadd_f32(float* dst, const float* a, const float* b, int n) {
  const int i = blockIdx.x * blockDim.x + threadIdx.x;
  if (i < n) dst[i] = a[i] + b[i];
}

__global__ void zero_f16(_Float16* p, size_t n) {
  const size_t i = (size_t)blockIdx.x * blockDim.x + threadIdx.x;
  if (i < n) p[i] = (_Float16)0.f;
}

__global__ void zero_f32(float* p, size_t n) {
  const size_t i = (size_t)blockIdx.x * blockDim.x + threadIdx.x;
  if (i < n) p[i] = 0.f;
}

// ---------------------------------------------------------------------------
extern "C" void kernel_launch(void* const* d_in, const int* in_sizes, int n_in,
                              void* d_out, int out_size, void* d_ws, size_t ws_size,
                              hipStream_t stream) {
  constexpr int B = 1024, T = 50, EIN = 96, EH = 128, G4 = 512;
  constexpr int CDEH = 64, MLPH = 128, CH = 33, N2 = CH * CDEH /*2112*/, NZ = 10000;
  constexpr int NSTEP = 16;

  const float* hist   = (const float*)d_in[0];
  const float* coeffs = (const float*)d_in[1];
  const float* Wih0 = (const float*)d_in[3];
  const float* Whh0 = (const float*)d_in[4];
  const float* bih0 = (const float*)d_in[5];
  const float* bhh0 = (const float*)d_in[6];
  const float* Wih1 = (const float*)d_in[7];
  const float* Whh1 = (const float*)d_in[8];
  const float* bih1 = (const float*)d_in[9];
  const float* bhh1 = (const float*)d_in[10];
  const float* Wmap = (const float*)d_in[11];
  const float* bmap = (const float*)d_in[12];
  const float* gamma = (const float*)d_in[13];
  const float* beta  = (const float*)d_in[14];
  const float* W1p = (const float*)d_in[15];
  const float* b1p = (const float*)d_in[16];
  const float* W2p = (const float*)d_in[17];
  const float* b2p = (const float*)d_in[18];
  const float* Wpred = (const float*)d_in[19];
  const float* bpred = (const float*)d_in[20];

  char* wp = (char*)d_ws;
  auto alloc = [&](size_t bytes) -> char* {
    char* r = wp;
    wp += (bytes + 255) & ~(size_t)255;
    return r;
  };

  _Float16* hist16  = (_Float16*)alloc((size_t)B * T * EIN * 2);
  _Float16* Wih0_16 = (_Float16*)alloc((size_t)G4 * EIN * 2);
  _Float16* Whh0_16 = (_Float16*)alloc((size_t)G4 * EH * 2);
  _Float16* Wih1_16 = (_Float16*)alloc((size_t)G4 * EH * 2);
  _Float16* Whh1_16 = (_Float16*)alloc((size_t)G4 * EH * 2);
  _Float16* Wmap16  = (_Float16*)alloc((size_t)CDEH * EH * 2);
  _Float16* W1_16   = (_Float16*)alloc((size_t)MLPH * CDEH * 2);
  _Float16* W2_16   = (_Float16*)alloc((size_t)N2 * MLPH * 2);
  _Float16* Wpred16 = (_Float16*)alloc((size_t)NZ * CDEH * 2);
  float*    bias0   = (float*)alloc((size_t)G4 * 4);
  float*    bias1   = (float*)alloc((size_t)G4 * 4);
  _Float16* gx16    = (_Float16*)alloc((size_t)B * T * G4 * 2);   // shared L0/L1
  _Float16* seq016  = (_Float16*)alloc((size_t)B * T * EH * 2);
  _Float16* h0a = (_Float16*)alloc((size_t)B * EH * 2);
  _Float16* h0b = (_Float16*)alloc((size_t)B * EH * 2);
  _Float16* h1a = (_Float16*)alloc((size_t)B * EH * 2);
  _Float16* h1b = (_Float16*)alloc((size_t)B * EH * 2);
  float* c0f = (float*)alloc((size_t)B * EH * 4);
  float* c1f = (float*)alloc((size_t)B * EH * 4);
  float* zf   = (float*)alloc((size_t)B * CDEH * 4);
  float* ytmp = (float*)alloc((size_t)B * CDEH * 4);
  float* kv[6];
  for (int i = 0; i < 6; ++i) kv[i] = (float*)alloc((size_t)B * CDEH * 4);
  _Float16* zn16 = (_Float16*)alloc((size_t)B * CDEH * 2);
  _Float16* a16  = (_Float16*)alloc((size_t)B * MLPH * 2);
  _Float16* z16  = (_Float16*)alloc((size_t)B * CDEH * 2);
  float* dxall = (float*)alloc((size_t)96 * B * CH * 4);

  auto cvt = [&](_Float16* dst, const float* src, size_t n) {
    cvt_f16<<<(unsigned)((n + 255) / 256), 256, 0, stream>>>(dst, src, n);
  };

  // --- weight / input conversions (f32 -> f16 WMMA operands) ---
  cvt(hist16, hist, (size_t)B * T * EIN);
  cvt(Wih0_16, Wih0, (size_t)G4 * EIN);
  cvt(Whh0_16, Whh0, (size_t)G4 * EH);
  cvt(Wih1_16, Wih1, (size_t)G4 * EH);
  cvt(Whh1_16, Whh1, (size_t)G4 * EH);
  cvt(Wmap16, Wmap, (size_t)CDEH * EH);
  cvt(W1_16, W1p, (size_t)MLPH * CDEH);
  cvt(W2_16, W2p, (size_t)N2 * MLPH);
  cvt(Wpred16, Wpred, (size_t)NZ * CDEH);
  vadd_f32<<<2, 256, 0, stream>>>(bias0, bih0, bhh0, G4);
  vadd_f32<<<2, 256, 0, stream>>>(bias1, bih1, bhh1, G4);

  // --- LSTM layer 0: hoisted input projection (one big WMMA GEMM) ---
  wmma_gemm<ACT_NONE, true>
      <<<(B * T / 64) * (G4 / 16), 32, 0, stream>>>(hist16, Wih0_16, bias0,
                                                    gx16, B * T, G4, EIN);
  zero_f16<<<(B * EH + 255) / 256, 256, 0, stream>>>(h0a, (size_t)B * EH);
  zero_f32<<<(B * EH + 255) / 256, 256, 0, stream>>>(c0f, (size_t)B * EH);
  {
    _Float16 *ha = h0a, *hb = h0b;
    for (int t = 0; t < T; ++t) {
      lstm_step<<<dim3(B / 16, EH / 16), 32, 0, stream>>>(
          ha, Whh0_16, gx16 + (size_t)t * G4, (size_t)T * G4, c0f, hb,
          seq016 + (size_t)t * EH, (size_t)T * EH);
      _Float16* tmp = ha; ha = hb; hb = tmp;
    }
  }

  // --- LSTM layer 1: hoisted input projection over seq0, then recurrence ---
  wmma_gemm<ACT_NONE, true>
      <<<(B * T / 64) * (G4 / 16), 32, 0, stream>>>(seq016, Wih1_16, bias1,
                                                    gx16, B * T, G4, EH);
  zero_f16<<<(B * EH + 255) / 256, 256, 0, stream>>>(h1a, (size_t)B * EH);
  zero_f32<<<(B * EH + 255) / 256, 256, 0, stream>>>(c1f, (size_t)B * EH);
  _Float16* ctx16 = h1a;
  {
    _Float16 *ha = h1a, *hb = h1b;
    for (int t = 0; t < T; ++t) {
      lstm_step<<<dim3(B / 16, EH / 16), 32, 0, stream>>>(
          ha, Whh1_16, gx16 + (size_t)t * G4, (size_t)T * G4, c1f, hb,
          (_Float16*)nullptr, 0);
      _Float16* tmp = ha; ha = hb; hb = tmp;
    }
    ctx16 = ha;  // final hidden state of layer 1
  }

  // --- map: z0 = tanh(ctx @ Wmap^T + bmap) ---
  wmma_gemm<ACT_TANH, false>
      <<<(B / 64) * (CDEH / 16), 32, 0, stream>>>(ctx16, Wmap16, bmap, zf, B,
                                                  CDEH, EH);

  // --- dXdt at all 96 RK stage times ---
  compute_dx<<<(96 * B * CH + 255) / 256, 256, 0, stream>>>(coeffs, dxall,
                                                            96 * B * CH);

  // --- Dormand-Prince RK45, 16 steps x 6 vf evals ---
  auto vf = [&](const float* zin, float* kout, int ti) {
    layernorm64<<<B, 64, 0, stream>>>(zin, gamma, beta, zn16);
    wmma_gemm<ACT_RELU, true>
        <<<(B / 64) * (MLPH / 16), 32, 0, stream>>>(zn16, W1_16, b1p, a16, B,
                                                    MLPH, CDEH);
    vf_mlp2<<<B / 16, 128, 0, stream>>>(a16, W2_16, b2p,
                                        dxall + (size_t)ti * B * CH, kout);
  };
  auto axpy = [&](float* out, const float* p1, float c1, const float* p2,
                  float c2, const float* p3, float c3, const float* p4,
                  float c4, const float* p5, float c5, const float* p6,
                  float c6) {
    axpy6<<<(B * CDEH + 255) / 256, 256, 0, stream>>>(
        out, zf, p1, c1, p2, c2, p3, c3, p4, c4, p5, c5, p6, c6, B * CDEH);
  };

  const float dt = 1.f / (float)NSTEP;
  for (int s = 0; s < NSTEP; ++s) {
    const int ti = s * 6;
    vf(zf, kv[0], ti + 0);
    axpy(ytmp, kv[0], dt / 5.f, zf, 0, zf, 0, zf, 0, zf, 0, zf, 0);
    vf(ytmp, kv[1], ti + 1);
    axpy(ytmp, kv[0], dt * 3.f / 40.f, kv[1], dt * 9.f / 40.f, zf, 0, zf, 0, zf, 0, zf, 0);
    vf(ytmp, kv[2], ti + 2);
    axpy(ytmp, kv[0], dt * 44.f / 45.f, kv[1], dt * -56.f / 15.f,
         kv[2], dt * 32.f / 9.f, zf, 0, zf, 0, zf, 0);
    vf(ytmp, kv[3], ti + 3);
    axpy(ytmp, kv[0], dt * 19372.f / 6561.f, kv[1], dt * -25360.f / 2187.f,
         kv[2], dt * 64448.f / 6561.f, kv[3], dt * -212.f / 729.f, zf, 0, zf, 0);
    vf(ytmp, kv[4], ti + 4);
    axpy(ytmp, kv[0], dt * 9017.f / 3168.f, kv[1], dt * -355.f / 33.f,
         kv[2], dt * 46732.f / 5247.f, kv[3], dt * 49.f / 176.f,
         kv[4], dt * -5103.f / 18656.f, zf, 0);
    vf(ytmp, kv[5], ti + 5);
    axpy(zf, kv[0], dt * 35.f / 384.f, kv[2], dt * 500.f / 1113.f,
         kv[3], dt * 125.f / 192.f, kv[4], dt * -2187.f / 6784.f,
         kv[5], dt * 11.f / 84.f, zf, 0);  // in-place z update
  }

  // --- prediction: out = z @ Wpred^T + bpred (1024 x 10000, K=64) ---
  cvt(z16, zf, (size_t)B * CDEH);
  wmma_gemm<ACT_NONE, false>
      <<<(B / 64) * (NZ / 16), 32, 0, stream>>>(z16, Wpred16, bpred,
                                                (float*)d_out, B, NZ, CDEH);
}
